// SentimentRNN_6949257085411
// MI455X (gfx1250) — matrix operand order repaired and checked by reading, not compiled, not run on hardware
//
#include <hip/hip_runtime.h>
#include <math.h>

typedef __bf16 bf16_t;
typedef __bf16 v16bf __attribute__((ext_vector_type(16)));
typedef float  v8f   __attribute__((ext_vector_type(8)));
typedef unsigned int u32x4 __attribute__((ext_vector_type(4)));
typedef int          i32x4 __attribute__((ext_vector_type(4)));
typedef int          i32x8 __attribute__((ext_vector_type(8)));

#if defined(__has_builtin)
#if __has_builtin(__builtin_amdgcn_tensor_load_to_lds) && \
    __has_builtin(__builtin_amdgcn_s_wait_tensorcnt)
#define USE_TDM 1
#else
#define USE_TDM 0
#endif
#else
#define USE_TDM 0
#endif

#define T_STEPS 512
#define BATCH   256
#define EDIM    512
#define HDIM    1024
#define KTOT    (EDIM + HDIM)   // 1536
#define LDS_PAD 8               // row stride 40 bf16 = 80B (16B aligned)

// ---------------------------------------------------------------------------
// Pack W_xh [H,E] and W_hh [H,H] (f32) into one bf16 matrix Wc [H][KTOT] so
// the step GEMM has a single contiguous K. 3 MB -> L2-resident for 512 steps.
// ---------------------------------------------------------------------------
__global__ void prep_weights(const float* __restrict__ Wxh,
                             const float* __restrict__ Whh,
                             bf16_t* __restrict__ Wc) {
  const int total = HDIM * KTOT;
  for (int i = blockIdx.x * blockDim.x + threadIdx.x; i < total;
       i += gridDim.x * blockDim.x) {
    const int n = i / KTOT;
    const int k = i - n * KTOT;
    const float v = (k < EDIM) ? Wxh[n * EDIM + k] : Whh[n * HDIM + (k - EDIM)];
    Wc[i] = (bf16_t)v;
  }
}

__global__ void zero_h(bf16_t* __restrict__ h) {
  const int i = blockIdx.x * blockDim.x + threadIdx.x;
  if (i < BATCH * HDIM) h[i] = (bf16_t)0.0f;
}

// ---------------------------------------------------------------------------
// TDM: DMA one 64-row x 32-K bf16 weight tile from Wc (row stride KTOT) into
// LDS, inserting 16B of pad after every 64B row so LDS rows land at the same
// 80B stride the ds_load_b128 fragment reads expect.
//   D# group0: count=1 | lds_addr | global_addr[56:0] | type=2
//   D# group1: data_size=1(2B), pad_enable, pad_interval=3 (16 DW = 64B),
//              pad_amount=3 (4 DW = 16B), tensor 1536x1024, tile 32x64,
//              dim0 stride 1536
// ---------------------------------------------------------------------------
__device__ __forceinline__ void tdm_load_b_tile(const bf16_t* gsrc,
                                                unsigned lds_off) {
#if USE_TDM
  const unsigned long long ga = (unsigned long long)(uintptr_t)gsrc;
  u32x4 g0;
  g0[0] = 1u;                                            // count=1 (valid)
  g0[1] = lds_off;                                       // lds_addr
  g0[2] = (unsigned)(ga & 0xFFFFFFFFu);                  // global_addr[31:0]
  g0[3] = (unsigned)((ga >> 32) & 0x01FFFFFFu) | (2u << 30);  // [56:32]|type=2
  i32x8 g1;
  g1[0] = (1 << 16) | (1 << 20) | (3 << 22) | (3 << 25); // sz=2B,pad,int,amt
  g1[1] = (int)((unsigned)KTOT << 16);                   // tensor_dim0[15:0]
  g1[2] = (int)((unsigned)HDIM << 16);                   // tensor_dim1[15:0]
  g1[3] = (int)(32u << 16);                              // tile_dim0 = 32
  g1[4] = 64;                                            // tile_dim1 = 64
  g1[5] = KTOT;                                          // dim0 stride
  g1[6] = 0;
  g1[7] = 0;
  const i32x4 z4 = {0, 0, 0, 0};
  const i32x8 z8 = {0, 0, 0, 0, 0, 0, 0, 0};
  // amdgpu-toolchain (clang-23) 6-arg form: (g0, g1, g2, g3, g4, cpol)
  __builtin_amdgcn_tensor_load_to_lds(g0, g1, z4, z4, z8, 0);
#else
  (void)gsrc; (void)lds_off;
#endif
}

// ---------------------------------------------------------------------------
// One recurrence step: h_out = tanh([emb[X_t] | h_in] * Wc^T + b_hh)
//   grid  = (HDIM/64, BATCH/128), block = 128 threads (4 x wave32)
//   wave w owns a 32-row strip of the 128x64 block: 2 A-frags x 4 B-frags
//   => 8 WMMAs per 32-wide K step, B fragments reused across both strips.
// ---------------------------------------------------------------------------
__global__ void __launch_bounds__(128)
rnn_step(const int* __restrict__ Xt,       // [BATCH] tokens at step t
         const float* __restrict__ emb,    // [V][EDIM] f32
         const bf16_t* __restrict__ Wc,    // [HDIM][KTOT] bf16
         const float* __restrict__ bhh,    // [HDIM]
         const bf16_t* __restrict__ hin,   // [BATCH][HDIM] bf16
         bf16_t* __restrict__ hout) {      // [BATCH][HDIM] bf16
  __shared__ __align__(16) bf16_t Alds[128][32 + LDS_PAD];
  __shared__ __align__(16) bf16_t Blds[64][32 + LDS_PAD];
  __shared__ int toks[128];

  const int tid  = threadIdx.x;
  const int lane = tid & 31;
  const int wm   = tid >> 5;            // wave id 0..3 -> 32-row strip
  const int n0   = blockIdx.x * 64;     // output-column (H) block
  const int bb0  = blockIdx.y * 128;    // batch-row block
  const int rm   = wm * 32;

  if (tid < 128) toks[tid] = Xt[bb0 + tid];
  __syncthreads();

#if USE_TDM
  const unsigned blds_off = (unsigned)(unsigned long long)
      (__attribute__((address_space(3))) char*)(&Blds[0][0]);
#endif

  v8f acc[2][4] = {};                   // 8 tiles of 16x16 f32

  const int c  = lane & 15;
  const int hh = lane >> 4;

  for (int k0 = 0; k0 < KTOT; k0 += 32) {
    const bool embPhase = (k0 < EDIM);
    // ---- B tile: TDM DMA (wave 0) or manual copy fallback ----
#if USE_TDM
    if (wm == 0) tdm_load_b_tile(Wc + (size_t)n0 * KTOT + k0, blds_off);
#else
    {
      const int n  = tid >> 1;
      const int kc = (tid & 1) * 16;
      const bf16_t* src = Wc + (size_t)(n0 + n) * KTOT + k0 + kc;
      #pragma unroll
      for (int j = 0; j < 16; ++j) Blds[n][kc + j] = src[j];
      if (k0 + 32 < KTOT) __builtin_prefetch(src + 32, 0, 1);
    }
#endif
    // ---- stage A block: 128 rows x 32 K ----
    if (embPhase) {
      #pragma unroll
      for (int it = 0; it < 4; ++it) {
        const int idx = tid + it * 128;       // 0..511
        const int r   = idx >> 2;
        const int kc  = (idx & 3) * 8;
        const float* src = emb + (size_t)toks[r] * EDIM + (k0 + kc);
        #pragma unroll
        for (int j = 0; j < 8; ++j) Alds[r][kc + j] = (bf16_t)src[j];
      }
    } else {
      #pragma unroll
      for (int it = 0; it < 4; ++it) {
        const int idx = tid + it * 128;
        const int r   = idx >> 2;
        const int kc  = (idx & 3) * 8;
        const bf16_t* src = hin + (size_t)(bb0 + r) * HDIM + (k0 - EDIM) + kc;
        #pragma unroll
        for (int j = 0; j < 8; ++j) Alds[r][kc + j] = src[j];
      }
    }
#if USE_TDM
    if (wm == 0) __builtin_amdgcn_s_wait_tensorcnt(0);
#endif
    __syncthreads();

    // ---- fragments per CDNA5 ISA 7.12.2 layouts ----
    // A 16x32 bf16: lane<16 row=lane holds K {0..7,16..23};
    //               lane>=16 row=lane-16 holds K {8..15,24..31}
    v16bf a[2];
    #pragma unroll
    for (int s = 0; s < 2; ++s) {
      const int r = rm + s * 16 + c;
      #pragma unroll
      for (int j = 0; j < 8; ++j) {
        a[s][j]     = Alds[r][hh * 8 + j];        // 16B contig -> ds b128
        a[s][8 + j] = Alds[r][16 + hh * 8 + j];   // 16B contig -> ds b128
      }
    }
    // B 32x16 bf16: lane<16 col=lane K {0..7,16..23}; lane>=16 col=lane-16 K {8..15,24..31}
    v16bf bf[4];
    #pragma unroll
    for (int tc = 0; tc < 4; ++tc)
      #pragma unroll
      for (int j = 0; j < 8; ++j) {
        bf[tc][j]     = Blds[tc * 16 + c][hh * 8 + j];
        bf[tc][8 + j] = Blds[tc * 16 + c][16 + hh * 8 + j];
      }

    #pragma unroll
    for (int s = 0; s < 2; ++s) {
      acc[s][0] = __builtin_amdgcn_wmma_f32_16x16x32_bf16(
          false, a[s], false, bf[0], (short)0, acc[s][0], false, false);
      acc[s][1] = __builtin_amdgcn_wmma_f32_16x16x32_bf16(
          false, a[s], false, bf[1], (short)0, acc[s][1], false, false);
      acc[s][2] = __builtin_amdgcn_wmma_f32_16x16x32_bf16(
          false, a[s], false, bf[2], (short)0, acc[s][2], false, false);
      acc[s][3] = __builtin_amdgcn_wmma_f32_16x16x32_bf16(
          false, a[s], false, bf[3], (short)0, acc[s][3], false, false);
    }
    __syncthreads();
  }

  // ---- epilogue: D layout => M = v + 8*(lane>=16), N = lane&15 ----
  #pragma unroll
  for (int tc = 0; tc < 4; ++tc) {
    const int   gc   = n0 + tc * 16 + c;
    const float bias = bhh[gc];
    #pragma unroll
    for (int s = 0; s < 2; ++s) {
      #pragma unroll
      for (int v = 0; v < 8; ++v) {
        const int gr = bb0 + rm + s * 16 + v + 8 * hh;
        hout[(size_t)gr * HDIM + gc] = (bf16_t)tanhf(acc[s][tc][v] + bias);
      }
    }
  }
}

// ---------------------------------------------------------------------------
// Head: yt = hT * W_hy^T + b_hy ; out = log_softmax(yt) over O=2
// ---------------------------------------------------------------------------
__global__ void finalize(const bf16_t* __restrict__ hT,
                         const float* __restrict__ Why,   // [2][HDIM]
                         const float* __restrict__ bhy,   // [2]
                         float* __restrict__ out) {       // [BATCH][2]
  const int b = blockIdx.x * blockDim.x + threadIdx.x;
  if (b >= BATCH) return;
  float y0 = bhy[0], y1 = bhy[1];
  for (int k = 0; k < HDIM; ++k) {
    const float hv = (float)hT[(size_t)b * HDIM + k];
    y0 += hv * Why[k];
    y1 += hv * Why[HDIM + k];
  }
  const float m   = fmaxf(y0, y1);
  const float l0  = y0 - m, l1 = y1 - m;
  const float lse = logf(expf(l0) + expf(l1));
  out[b * 2 + 0] = l0 - lse;
  out[b * 2 + 1] = l1 - lse;
}

// ---------------------------------------------------------------------------
extern "C" void kernel_launch(void* const* d_in, const int* in_sizes, int n_in,
                              void* d_out, int out_size, void* d_ws, size_t ws_size,
                              hipStream_t stream) {
  (void)in_sizes; (void)n_in; (void)out_size; (void)ws_size;
  const int*   X   = (const int*)d_in[0];    // [T,B]
  const float* emb = (const float*)d_in[1];  // [V,E]
  const float* Wxh = (const float*)d_in[2];  // [H,E]
  const float* Whh = (const float*)d_in[3];  // [H,H]
  const float* bhh = (const float*)d_in[4];  // [H]
  const float* Why = (const float*)d_in[5];  // [2,H]
  const float* bhy = (const float*)d_in[6];  // [2]
  float* out = (float*)d_out;

  char* ws = (char*)d_ws;                              // 4 MB total
  bf16_t* Wc = (bf16_t*)ws;                            // 3,145,728 B
  bf16_t* hA = (bf16_t*)(ws + (size_t)HDIM * KTOT * sizeof(bf16_t));
  bf16_t* hB = hA + (size_t)BATCH * HDIM;              // 524,288 B each

  prep_weights<<<1024, 256, 0, stream>>>(Wxh, Whh, Wc);
  zero_h<<<(BATCH * HDIM) / 256, 256, 0, stream>>>(hA);

  bf16_t* bufs[2] = {hA, hB};
  for (int t = 0; t < T_STEPS; ++t) {
    rnn_step<<<dim3(HDIM / 64, BATCH / 128), 128, 0, stream>>>(
        X + (size_t)t * BATCH, emb, Wc, bhh, bufs[t & 1], bufs[(t + 1) & 1]);
  }
  // after 512 steps (even), final h is back in buffer 0
  finalize<<<1, 256, 0, stream>>>(bufs[0], Why, bhy, out);
}
